// GRU_70377334112752
// MI455X (gfx1250) — matrix operand order repaired and checked by reading, not compiled, not run on hardware
//
#include <hip/hip_runtime.h>

typedef __attribute__((ext_vector_type(8)))  float   v8f;
typedef __attribute__((ext_vector_type(8)))  __bf16  v8bf;
typedef __attribute__((ext_vector_type(16))) __bf16  v16bf;

#define B_   64
#define N_   1024
#define T_   512
#define BT_  (B_ * T_)     // 32768
#define NG_  (3 * N_)      // 3072
#define RBLOCKS 64

// ---------------------------------------------------------------------------
// WMMA helpers (CDNA5 wave32, 16x16x32 bf16 -> f32)
// ---------------------------------------------------------------------------
__device__ __forceinline__ v8f wmma_bf16(v16bf a, v16bf b, v8f c) {
    return __builtin_amdgcn_wmma_f32_16x16x32_bf16(
        /*neg_a=*/false, a, /*neg_b=*/false, b,
        /*c_mod=*/(short)0, c, /*reuse_a=*/false, /*reuse_b=*/false);
}

// A fragment: 16x32 (MxK) row-major source. Lane l holds row l%16,
// K = k0 + (l<16 ? 0..15 : 16..31), contiguous.
__device__ __forceinline__ v16bf load_afrag(const __bf16* A, int lda,
                                            int row0, int k0, int lane) {
    int r = row0 + (lane & 15);
    const __bf16* p = A + (size_t)r * lda + k0 + ((lane & 16) ? 16 : 0);
    v8bf lo = *(const v8bf*)(p);
    v8bf hi = *(const v8bf*)(p + 8);
    v16bf a;
#pragma unroll
    for (int i = 0; i < 8; ++i) { a[i] = lo[i]; a[i + 8] = hi[i]; }
    return a;
}

// B fragment: 32x16 (KxN) from column-major (transposed) weights Wt[col][k].
// Lane l holds column l%16, K = k0 + (l<16 ? 0..15 : 16..31), contiguous.
__device__ __forceinline__ v16bf load_bfrag(const __bf16* Wt, int col0,
                                            int k0, int lane) {
    int c = col0 + (lane & 15);
    const __bf16* p = Wt + (size_t)c * N_ + k0 + ((lane & 16) ? 16 : 0);
    return *(const v16bf*)p;
}

__device__ __forceinline__ float sigmoidf_(float x) {
    return 1.0f / (1.0f + __expf(-x));
}

// CDNA5 async global->LDS copy (16 bytes per lane), tracked by ASYNCcnt.
__device__ __forceinline__ void async_g2l_b128(unsigned lds_byte_off,
                                               const void* gptr) {
    asm volatile("global_load_async_to_lds_b128 %0, %1, off"
                 :: "v"(lds_byte_off), "v"(gptr)
                 : "memory");
}
__device__ __forceinline__ void wait_async0() {
    asm volatile("s_wait_asynccnt 0" ::: "memory");
}

// ---------------------------------------------------------------------------
// Init: zero h double-buffers (bf16 + f32) and grid-barrier state
// ---------------------------------------------------------------------------
__global__ void k_init(__bf16* h0, __bf16* h1, float* h0f, float* h1f,
                       unsigned* bar) {
    int i = blockIdx.x * blockDim.x + threadIdx.x;
    if (i < B_ * N_) {
        h0[i] = (__bf16)0.0f; h1[i] = (__bf16)0.0f;
        h0f[i] = 0.0f;        h1f[i] = 0.0f;
    }
    if (i < 4) bar[i] = 0u;
}

// ---------------------------------------------------------------------------
// flow_x [B, N, T] f32  ->  Xbf [B*T, N] bf16   (LDS-tiled transpose)
// ---------------------------------------------------------------------------
__global__ void k_transpose_x(const float* __restrict__ flow,
                              __bf16* __restrict__ Xbf) {
    __shared__ float tile[32][33];
    int t0 = blockIdx.x * 32, n0 = blockIdx.y * 32, b = blockIdx.z;
    const float* src = flow + (size_t)b * N_ * T_;
#pragma unroll
    for (int i = 0; i < 4; ++i) {
        int n = n0 + threadIdx.y + i * 8;
        tile[threadIdx.y + i * 8][threadIdx.x] =
            src[(size_t)n * T_ + t0 + threadIdx.x];
    }
    __syncthreads();
    __bf16* dst = Xbf + (size_t)b * T_ * N_;
#pragma unroll
    for (int i = 0; i < 4; ++i) {
        int t = t0 + threadIdx.y + i * 8;
        dst[(size_t)t * N_ + n0 + threadIdx.x] =
            (__bf16)tile[threadIdx.x][threadIdx.y + i * 8];
    }
}

// ---------------------------------------------------------------------------
// W1 [1024,2048], W2 [1024,1024] f32 -> Wt [3][1024(col)][1024(k)] bf16
// ---------------------------------------------------------------------------
__global__ void k_transpose_w(const float* __restrict__ W1,
                              const float* __restrict__ W2,
                              __bf16* __restrict__ Wt) {
    __shared__ float tile[32][33];
    int g  = blockIdx.z;
    int j0 = blockIdx.x * 32, k0 = blockIdx.y * 32;
    const float* src; int ld, coff;
    if (g == 0)      { src = W1; ld = 2048; coff = 0;    }
    else if (g == 1) { src = W1; ld = 2048; coff = 1024; }
    else             { src = W2; ld = 1024; coff = 0;    }
#pragma unroll
    for (int i = 0; i < 4; ++i) {
        int k = k0 + threadIdx.y + i * 8;
        tile[threadIdx.y + i * 8][threadIdx.x] =
            src[(size_t)k * ld + coff + j0 + threadIdx.x];
    }
    __syncthreads();
    __bf16* dst = Wt + (size_t)g * N_ * N_;
#pragma unroll
    for (int i = 0; i < 4; ++i) {
        int j = j0 + threadIdx.y + i * 8;
        dst[(size_t)j * N_ + k0 + threadIdx.x] =
            (__bf16)tile[threadIdx.x][threadIdx.y + i * 8];
    }
}

// ---------------------------------------------------------------------------
// Precompute: Xp[bt, 3072] = Xbf[bt, :] @ W + bias   (bf16 WMMA, f32 acc)
// grid = (48, 512), 128 threads (4 waves), block tile 64x64
// ---------------------------------------------------------------------------
__global__ void __launch_bounds__(128)
k_gemm_x(const __bf16* __restrict__ Xbf, const __bf16* __restrict__ Wt,
         const float* __restrict__ b1, const float* __restrict__ b2,
         __bf16* __restrict__ Xp) {
    int lane = threadIdx.x & 31;
    int wave = threadIdx.x >> 5;
    int row0 = blockIdx.y * 64 + wave * 16;
    int col0 = blockIdx.x * 64;

    v8f acc[4] = {};
    for (int k0 = 0; k0 < N_; k0 += 32) {
        v16bf a = load_afrag(Xbf, N_, row0, k0, lane);
#pragma unroll
        for (int n = 0; n < 4; ++n) {
            v16bf bfrag = load_bfrag(Wt, col0 + n * 16, k0, lane);
            acc[n] = wmma_bf16(a, bfrag, acc[n]);
        }
    }
    int moff = (lane & 16) ? 8 : 0;
#pragma unroll
    for (int n = 0; n < 4; ++n) {
        int c = col0 + n * 16 + (lane & 15);
        float bias = (c < 2048) ? b1[c] : b2[c - 2048];
#pragma unroll
        for (int e = 0; e < 8; ++e) {
            int r = row0 + e + moff;
            Xp[(size_t)r * NG_ + c] = (__bf16)(acc[n][e] + bias);
        }
    }
}

// ---------------------------------------------------------------------------
// Grid-wide barrier (generation counter in device memory, agent scope)
// ---------------------------------------------------------------------------
__device__ __forceinline__ void grid_sync(unsigned* bar, unsigned nb) {
    __threadfence();
    __syncthreads();
    if (threadIdx.x == 0) {
        unsigned* cnt = bar;
        unsigned* gen = bar + 1;
        unsigned g = __hip_atomic_load(gen, __ATOMIC_RELAXED,
                                       __HIP_MEMORY_SCOPE_AGENT);
        unsigned prev = __hip_atomic_fetch_add(cnt, 1u, __ATOMIC_ACQ_REL,
                                               __HIP_MEMORY_SCOPE_AGENT);
        if (prev == nb - 1) {
            __hip_atomic_store(cnt, 0u, __ATOMIC_RELAXED,
                               __HIP_MEMORY_SCOPE_AGENT);
            __hip_atomic_store(gen, g + 1, __ATOMIC_RELEASE,
                               __HIP_MEMORY_SCOPE_AGENT);
        } else {
            while (__hip_atomic_load(gen, __ATOMIC_ACQUIRE,
                                     __HIP_MEMORY_SCOPE_AGENT) == g) {
                __builtin_amdgcn_s_sleep(1);
            }
        }
    }
    __syncthreads();
}

// ---------------------------------------------------------------------------
// Persistent recurrent scan: 64 blocks x 128 threads; block owns 16 columns,
// wave w owns batch rows 16w..16w+15. One grid barrier per timestep.
// Weight panels (3 gates x 16 cols x 1024 k = 96 KB bf16) are staged ONCE
// into LDS via async global->LDS copies and reused for all 512 steps.
// ---------------------------------------------------------------------------
__global__ void __launch_bounds__(128)
k_gru_scan(const __bf16* __restrict__ Wt, const float* __restrict__ b1,
           const float* __restrict__ b2, const __bf16* __restrict__ Xp,
           __bf16* h0buf, __bf16* h1buf, float* h0f, float* h1f,
           unsigned* bar, float* __restrict__ out) {
    __shared__ __bf16 sW[3 * 16 * N_];   // 96 KB of WGP LDS

    int lane = threadIdx.x & 31;
    int wave = threadIdx.x >> 5;
    int col0 = blockIdx.x * 16;
    int row0 = wave * 16;
    int cl   = col0 + (lane & 15);
    int moff = (lane & 16) ? 8 : 0;

    // ---- stage weight panels into LDS (async, 16B per lane per op) ----
    for (int i = threadIdx.x; i < (3 * 16 * N_) / 8; i += 128) {
        int e = i * 8;                 // bf16 element index into sW
        int r = e >> 10;               // local row 0..47
        int k = e & (N_ - 1);
        int g = r >> 4, c = r & 15;
        const __bf16* gp = Wt + ((size_t)(g * N_ + col0 + c) * N_ + k);
        async_g2l_b128((unsigned)(uintptr_t)(sW + e), gp);
    }
    wait_async0();
    __syncthreads();

    float bz = b1[cl], br = b1[1024 + cl], bh = b2[cl];

    for (int t = 0; t < T_; ++t) {
        const __bf16* hc  = (t & 1) ? h1buf : h0buf;
        __bf16*       hn  = (t & 1) ? h0buf : h1buf;
        const float*  hcf = (t & 1) ? h1f : h0f;
        float*        hnf = (t & 1) ? h0f : h1f;

        // Issue the per-step input-projection / h loads BEFORE the GEMM so
        // their latency hides under the 32 WMMA k-steps.
        float x1v[8], x2v[8], xhv[8], hpv[8];
#pragma unroll
        for (int e = 0; e < 8; ++e) {
            int m = row0 + e + moff;                    // batch index
            const __bf16* xp = Xp + ((size_t)m * T_ + t) * NG_;
            x1v[e] = (float)xp[cl];
            x2v[e] = (float)xp[N_ + cl];
            xhv[e] = (float)xp[2 * N_ + cl];
            hpv[e] = hcf[(size_t)m * N_ + cl];
        }

        v8f az = {}, ar = {}, ah = {};
        for (int k0 = 0; k0 < N_; k0 += 32) {
            v16bf a = load_afrag(hc, N_, row0, k0, lane);
            int koff = k0 + ((lane & 16) ? 16 : 0);
            v16bf bzf = *(const v16bf*)(sW + ((0 * 16 + (lane & 15)) * N_ + koff));
            v16bf brf = *(const v16bf*)(sW + ((1 * 16 + (lane & 15)) * N_ + koff));
            v16bf bhf = *(const v16bf*)(sW + ((2 * 16 + (lane & 15)) * N_ + koff));
            az = wmma_bf16(a, bzf, az);
            ar = wmma_bf16(a, brf, ar);
            ah = wmma_bf16(a, bhf, ah);
        }

#pragma unroll
        for (int e = 0; e < 8; ++e) {
            int m = row0 + e + moff;
            float z    = sigmoidf_(x1v[e] + az[e] + bz);
            float r    = sigmoidf_(x2v[e] + ar[e] + br);
            float hhat = tanhf(xhv[e] + r * (ah[e] + bh));
            float hnew = z * hpv[e] + (1.0f - z) * hhat;
            hn[(size_t)m * N_ + cl]  = (__bf16)hnew;
            hnf[(size_t)m * N_ + cl] = hnew;
            if (t == T_ - 1) out[(size_t)m * N_ + cl] = hnew;
            // Warm next step's Xp row while we sit in the grid barrier.
            if (t + 1 < T_)
                __builtin_prefetch(Xp + ((size_t)m * T_ + (t + 1)) * NG_ + cl, 0, 1);
        }
        grid_sync(bar, RBLOCKS);
    }
}

// ---------------------------------------------------------------------------
// Launch
// ---------------------------------------------------------------------------
extern "C" void kernel_launch(void* const* d_in, const int* in_sizes, int n_in,
                              void* d_out, int out_size, void* d_ws,
                              size_t ws_size, hipStream_t stream) {
    (void)in_sizes; (void)n_in; (void)out_size; (void)ws_size;
    const float* flow = (const float*)d_in[0];
    const float* W1   = (const float*)d_in[1];
    const float* b1   = (const float*)d_in[2];
    const float* W2   = (const float*)d_in[3];
    const float* b2   = (const float*)d_in[4];
    float* out = (float*)d_out;

    char* ws = (char*)d_ws;
    size_t off = 0;
    __bf16* Xbf = (__bf16*)(ws + off); off += (size_t)BT_ * N_ * 2;    //  64 MB
    __bf16* Wt  = (__bf16*)(ws + off); off += (size_t)3 * N_ * N_ * 2; //   6 MB
    __bf16* Xp  = (__bf16*)(ws + off); off += (size_t)BT_ * NG_ * 2;   // 192 MB
    __bf16* h0  = (__bf16*)(ws + off); off += (size_t)B_ * N_ * 2;
    __bf16* h1  = (__bf16*)(ws + off); off += (size_t)B_ * N_ * 2;
    float*  h0f = (float*)(ws + off);  off += (size_t)B_ * N_ * 4;
    float*  h1f = (float*)(ws + off);  off += (size_t)B_ * N_ * 4;
    unsigned* bar = (unsigned*)(ws + off); off += 256;

    k_init<<<(B_ * N_ + 255) / 256, 256, 0, stream>>>(h0, h1, h0f, h1f, bar);
    k_transpose_x<<<dim3(T_ / 32, N_ / 32, B_), dim3(32, 8), 0, stream>>>(flow, Xbf);
    k_transpose_w<<<dim3(32, 32, 3), dim3(32, 8), 0, stream>>>(W1, W2, Wt);
    k_gemm_x<<<dim3(NG_ / 64, BT_ / 64), 128, 0, stream>>>(Xbf, Wt, b1, b2, Xp);
    k_gru_scan<<<dim3(N_ / 16), 128, 0, stream>>>(Wt, b1, b2, Xp, h0, h1,
                                                  h0f, h1f, bar, out);
}
